// OptimizationCanonizationNetwork_61512521613519
// MI455X (gfx1250) — compile-verified, matrix operand order checked by reading, not run on hardware
//
#include <hip/hip_runtime.h>
#include <hip/hip_bf16.h>

typedef __attribute__((ext_vector_type(16))) _Float16 v16h;
typedef __attribute__((ext_vector_type(8)))  _Float16 v8h;
typedef __attribute__((ext_vector_type(8)))  float    v8f;

#define BATCH   256
#define NPTS    784
#define HID     64
#define IMG     28
#define OCH     32
#define OWH     13
#define KGEMM   5408      // 32*13*13 = 169*32, divisible by 32
#define NOUT    10
#define NPAD    16
#define NITERS  5
#define LRHALF  0.05f     // LR * 0.5
#define CXY     14.0f

// ---------------------------------------------------------------------------
// CDNA5 async global->LDS copy (16B per active lane) + completion wait.
// Inline asm per bridge doc: portable across ROCm-7.2 / amdgpu-toolchain,
// whose tensor/async builtins differ in arity. The VDST operand is the
// wave-relative LDS byte offset == low 32 bits of the generic pointer.
// ---------------------------------------------------------------------------
__device__ __forceinline__ void async_copy_b128(void* lds_dst, const void* gsrc) {
    asm volatile("global_load_async_to_lds_b128 %0, %1, off"
                 :: "v"((unsigned)(unsigned long long)lds_dst),
                    "v"((unsigned long long)gsrc)
                 : "memory");
}
__device__ __forceinline__ void wait_async0() {
    asm volatile("s_wait_asynccnt 0x0" ::: "memory");
}

// ---------------------------------------------------------------------------
// block-wide f32 sum over 256 threads; returns result to all threads
// ---------------------------------------------------------------------------
__device__ inline float block_reduce256(float v, float* red, int tid) {
    red[tid] = v;
    __syncthreads();
    #pragma unroll
    for (int s = 128; s > 0; s >>= 1) {
        if (tid < s) red[tid] += red[tid + s];
        __syncthreads();
    }
    float r = red[0];
    __syncthreads();
    return r;
}

// ---------------------------------------------------------------------------
// Kernel 1: per-batch angle optimization (5 fwd-mode GD iterations).
// Rank-3 structure folds the 64x64 matmuls into M2 = w_start@w_id and
// M3 = w_start@w_pool. Points are staged to LDS with async-to-LDS copies
// that overlap the folded-weight precompute (~12K FMAs of latency cover).
// ---------------------------------------------------------------------------
__global__ __launch_bounds__(256)
void k_angle(const float* __restrict__ points,
             const float* __restrict__ w_start, const float* __restrict__ b_start,
             const float* __restrict__ w_id,    const float* __restrict__ b_id,
             const float* __restrict__ w_pool,  const float* __restrict__ b_pool,
             const float* __restrict__ w_out,
             float* __restrict__ rot)
{
    __shared__ __align__(16) float sP[NPTS * 3];          // 9408 B = 588 x 16B
    __shared__ float sM2[3][HID], sM3[3][HID];
    __shared__ float sC0[HID];             // b_start@w_id + b_id
    __shared__ float sPB[HID];             // N*(b_start@w_pool) + b_pool
    __shared__ float sWo[HID];
    __shared__ float sW0[HID], sW1[HID];   // M2 rows 0/1 pre-scaled by w_out
    __shared__ float sC1[HID], sPD[HID];   // per-iteration constants
    __shared__ float red[256];

    const int b = blockIdx.x, tid = threadIdx.x;
    const char* gp = (const char*)(points + (size_t)b * (NPTS * 3));

    // --- async stage points -> LDS (overlaps with the precompute below) ---
    #pragma unroll
    for (int i = 0; i < 3; ++i) {
        int chunk = tid + 256 * i;
        if (chunk < (NPTS * 3 * 4) / 16)
            async_copy_b128((char*)sP + chunk * 16, gp + chunk * 16);
    }

    // folded weights: M2[r][k] = sum_j w_start[r][j] * w_id[j][k], same for M3
    if (tid < 192) {
        int r = tid >> 6, k = tid & 63;
        float m2 = 0.f, m3 = 0.f;
        for (int j = 0; j < HID; ++j) {
            float ws = w_start[r * HID + j];
            m2 = fmaf(ws, w_id[j * HID + k],   m2);
            m3 = fmaf(ws, w_pool[j * HID + k], m3);
        }
        sM2[r][k] = m2; sM3[r][k] = m3;
    }
    if (tid < HID) {
        int k = tid;
        float c0 = b_id[k], p0 = 0.f;
        for (int j = 0; j < HID; ++j) {
            c0 = fmaf(b_start[j], w_id[j * HID + k],   c0);
            p0 = fmaf(b_start[j], w_pool[j * HID + k], p0);
        }
        sC0[k] = c0;
        sPB[k] = (float)NPTS * p0 + b_pool[k];
        sWo[k] = w_out[k];
    }

    // d0/d1 = w_start rows 0/1 dotted with w_out (unmasked dh term)
    float d0 = 0.f, d1 = 0.f;
    for (int k = 0; k < HID; ++k) {
        d0 = fmaf(w_start[k],       w_out[k], d0);
        d1 = fmaf(w_start[HID + k], w_out[k], d1);
    }

    wait_async0();          // points resident in LDS for this wave
    __syncthreads();        // ... and for all waves

    if (tid < HID) {
        sW0[tid] = sM2[0][tid] * sWo[tid];
        sW1[tid] = sM2[1][tid] * sWo[tid];
    }

    // thread-private points (<=4 each) + local coordinate sums
    float px[4], py[4], pz[4];
    bool  pv[4];
    float lx = 0.f, ly = 0.f, lz = 0.f;
    #pragma unroll
    for (int i = 0; i < 4; ++i) {
        int n = tid + 256 * i;
        pv[i] = (n < NPTS);
        px[i] = pv[i] ? sP[n * 3 + 0] : 0.f;
        py[i] = pv[i] ? sP[n * 3 + 1] : 0.f;
        pz[i] = pv[i] ? sP[n * 3 + 2] : 0.f;
        lx += px[i]; ly += py[i]; lz += pz[i];
    }
    float Sx = block_reduce256(lx, red, tid);
    float Sy = block_reduce256(ly, red, tid);
    float Sz = block_reduce256(lz, red, tid);

    float theta = 0.f;
    for (int it = 0; it < NITERS; ++it) {
        float c = cosf(theta), s = sinf(theta);
        float Srx = c * Sx - s * Sy;
        float Sry = s * Sx + c * Sy;
        if (tid < HID) {
            int k = tid;
            float pooled = fmaf(Srx, sM3[0][k],
                           fmaf(Sry, sM3[1][k],
                           fmaf(Sz,  sM3[2][k], sPB[k])));
            sC1[k] = sC0[k] + pooled;
            sPD[k] = (Srx * sM3[1][k] - Sry * sM3[0][k]) * sWo[k];
        }
        __syncthreads();

        float g = 0.f;
        float rx[4], ry[4];
        #pragma unroll
        for (int i = 0; i < 4; ++i) {
            rx[i] = c * px[i] - s * py[i];
            ry[i] = s * px[i] + c * py[i];
        }
        for (int k = 0; k < HID; ++k) {
            float m0 = sM2[0][k], m1 = sM2[1][k], m2z = sM2[2][k];
            float c1 = sC1[k], w0 = sW0[k], w1 = sW1[k], pd = sPD[k];
            #pragma unroll
            for (int i = 0; i < 4; ++i) {
                if (pv[i]) {
                    float pre = fmaf(rx[i], m0, fmaf(ry[i], m1, fmaf(pz[i], m2z, c1)));
                    float tw  = fmaf(rx[i], w1, fmaf(-ry[i], w0, pd));
                    g += (pre > 0.f) ? tw : 0.f;
                }
            }
        }
        g = block_reduce256(g, red, tid);
        g += Srx * d1 - Sry * d0;        // unmasked +h tangent term
        theta -= LRHALF * g;
    }

    if (tid == 0) {
        float c = cosf(theta), s = sinf(theta);
        rot[b * 4 + 0] = c;
        rot[b * 4 + 1] = s;
        rot[b * 4 + 2] = (1.f - c) * CXY - s * CXY;
        rot[b * 4 + 3] = s * CXY + (1.f - c) * CXY;
    }
}

// ---------------------------------------------------------------------------
// Kernel 2: fused bilinear inverse-warp + 3x3/stride-2 conv + ReLU -> f16 reps
// Image + conv weights staged via async-to-LDS (276 x 16B chunks), overlapped
// with the per-batch rotation parameter fetch.
// ---------------------------------------------------------------------------
__device__ inline float sampleImg(const float* img, int y, int x) {
    if (x < 0 || x > IMG - 1 || y < 0 || y > IMG - 1) return 0.f;
    return img[y * IMG + x];
}

__global__ __launch_bounds__(256)
void k_warp_conv(const float* __restrict__ images,
                 const float* __restrict__ conv_w, const float* __restrict__ conv_b,
                 const float* __restrict__ rot,
                 _Float16* __restrict__ reps)
{
    __shared__ __align__(16) float sImg[NPTS];      // 3136 B = 196 x 16B
    __shared__ __align__(16) float sCW[OCH * 9];    // 1152 B =  72 x 16B
    __shared__ __align__(16) float sCB[OCH];        //  128 B =   8 x 16B
    __shared__ float sWarp[NPTS];

    const int b = blockIdx.x, tid = threadIdx.x;

    if (tid < 196)
        async_copy_b128((char*)sImg + tid * 16,
                        (const char*)(images + (size_t)b * NPTS) + tid * 16);
    else if (tid < 268)
        async_copy_b128((char*)sCW + (tid - 196) * 16,
                        (const char*)conv_w + (tid - 196) * 16);
    else if (tid < 276)
        async_copy_b128((char*)sCB + (tid - 268) * 16,
                        (const char*)conv_b + (tid - 268) * 16);

    // overlap: fetch per-batch affine params while the DMA runs
    float c  = rot[b * 4 + 0], s  = rot[b * 4 + 1];
    float t0 = rot[b * 4 + 2], t1 = rot[b * 4 + 3];
    // det(R) == 1, so Rinv = [[c,-s],[s,c]]; subtract Rinv@t
    float it0 = c * t0 - s * t1;
    float it1 = s * t0 + c * t1;

    wait_async0();
    __syncthreads();

    for (int p = tid; p < NPTS; p += 256) {
        float dx = (float)(p % IMG), dy = (float)(p / IMG);
        float sx = c * dx - s * dy - it0;
        float sy = s * dx + c * dy - it1;
        float x0f = floorf(sx), y0f = floorf(sy);
        float wx = sx - x0f, wy = sy - y0f;
        int x0 = (int)x0f, y0 = (int)y0f;
        float v00 = sampleImg(sImg, y0,     x0);
        float v01 = sampleImg(sImg, y0,     x0 + 1);
        float v10 = sampleImg(sImg, y0 + 1, x0);
        float v11 = sampleImg(sImg, y0 + 1, x0 + 1);
        sWarp[p] = v00 * (1.f - wx) * (1.f - wy) + v01 * wx * (1.f - wy)
                 + v10 * (1.f - wx) * wy         + v11 * wx * wy;
    }
    __syncthreads();

    for (int o = tid; o < KGEMM; o += 256) {
        int oc  = o / (OWH * OWH);
        int rem = o % (OWH * OWH);
        int oy  = rem / OWH, ox = rem % OWH;
        float acc = sCB[oc];
        #pragma unroll
        for (int ky = 0; ky < 3; ++ky)
            #pragma unroll
            for (int kx = 0; kx < 3; ++kx)
                acc = fmaf(sWarp[(2 * oy + ky) * IMG + (2 * ox + kx)],
                           sCW[oc * 9 + ky * 3 + kx], acc);
        reps[(size_t)b * KGEMM + o] = (_Float16)fmaxf(acc, 0.f);
    }
}

// ---------------------------------------------------------------------------
// Kernel 0: transpose + f16-convert + pad w_pred (5408x10) -> Bt[16][5408]
// ---------------------------------------------------------------------------
__global__ __launch_bounds__(256)
void k_prep_b(const float* __restrict__ w_pred, _Float16* __restrict__ Bt)
{
    int idx = blockIdx.x * 256 + threadIdx.x;   // idx = n*KGEMM + k
    if (idx >= NPAD * KGEMM) return;
    int n = idx / KGEMM, k = idx % KGEMM;
    float v = (n < NOUT) ? w_pred[k * NOUT + n] : 0.f;
    Bt[idx] = (_Float16)v;
}

// ---------------------------------------------------------------------------
// Kernel 3: classifier head GEMM (256 x 5408) @ (5408 x 16-padded) via
// v_wmma_f32_16x16x32_f16, software-pipelined: WMMA on tile t overlaps the
// b128 fragment loads of tile t+1 (one-deep double buffer), plus near-scope
// prefetch ~4 tiles ahead. Fragment layout per ISA 7.12.2: lanes 0-15 carry
// K {0..7,16..23}, lanes 16-31 carry K {8..15,24..31}; B mirrored (lane=col)
// via the [N][K] pre-transpose so every lane load is a contiguous b128.
// ---------------------------------------------------------------------------
__device__ __forceinline__ void load_frag(const _Float16* p, v16h& F) {
    v8h lo = *(const v8h*)(p);
    v8h hi = *(const v8h*)(p + 16);
    #pragma unroll
    for (int e = 0; e < 8; ++e) { F[e] = lo[e]; F[e + 8] = hi[e]; }
}

__global__ __launch_bounds__(32)
void k_gemm(const _Float16* __restrict__ reps, const _Float16* __restrict__ Bt,
            const float* __restrict__ b_pred, float* __restrict__ out)
{
    const int mtile   = blockIdx.x;
    const int lane    = threadIdx.x;
    const int halfsel = lane >> 4;     // 0: K{0-7,16-23}; 1: K{8-15,24-31}
    const int mr      = lane & 15;     // A row within tile / B column

    const _Float16* Ar = reps + (size_t)(mtile * 16 + mr) * KGEMM + halfsel * 8;
    const _Float16* Br = Bt   + (size_t)mr * KGEMM + halfsel * 8;

    v16h A0, Bv0;
    load_frag(Ar, A0);
    load_frag(Br, Bv0);

    v8f acc = {};
    for (int kk = 0; kk < KGEMM; kk += 32) {
        int nk = kk + 32;
        v16h A1 = A0, Bv1 = Bv0;
        if (nk < KGEMM) {
            int pf = (kk + 160 < KGEMM) ? kk + 160 : kk;   // clamped, in-buffer
            __builtin_prefetch(Ar + pf, 0, 3);             // near-scope prefetch
            __builtin_prefetch(Br + pf, 0, 3);
            load_frag(Ar + nk, A1);
            load_frag(Br + nk, Bv1);
        }
        acc = __builtin_amdgcn_wmma_f32_16x16x32_f16(
                  false, A0, false, Bv0, (short)0, acc, false, false);
        A0 = A1; Bv0 = Bv1;
    }

    // C layout: lanes 0-15 -> (M=r, N=lane); lanes 16-31 -> (M=8+r, N=lane-16)
    int n = mr;
    if (n < NOUT) {
        int mbase = mtile * 16 + halfsel * 8;
        #pragma unroll
        for (int r = 0; r < 8; ++r)
            out[(mbase + r) * NOUT + n] = acc[r] + b_pred[n];
    }
}

// ---------------------------------------------------------------------------
extern "C" void kernel_launch(void* const* d_in, const int* in_sizes, int n_in,
                              void* d_out, int out_size, void* d_ws, size_t ws_size,
                              hipStream_t stream)
{
    const float* images  = (const float*)d_in[0];
    const float* points  = (const float*)d_in[1];
    const float* w_start = (const float*)d_in[2];
    const float* b_start = (const float*)d_in[3];
    const float* w_id    = (const float*)d_in[4];
    const float* b_id    = (const float*)d_in[5];
    const float* w_pool  = (const float*)d_in[6];
    const float* b_pool  = (const float*)d_in[7];
    const float* w_out   = (const float*)d_in[8];
    // d_in[9] = b_out: constant offset, vanishes from the gradient and output
    const float* conv_w  = (const float*)d_in[10];
    const float* conv_b  = (const float*)d_in[11];
    const float* w_pred  = (const float*)d_in[12];
    const float* b_pred  = (const float*)d_in[13];
    float* out = (float*)d_out;

    char* ws = (char*)d_ws;
    float*    rot  = (float*)ws;                                   // 256*4 f32
    _Float16* Bt   = (_Float16*)(ws + 4096);                       // 16*5408 f16
    _Float16* reps = (_Float16*)(ws + 4096 + NPAD * KGEMM * 2);    // 256*5408 f16

    k_prep_b<<<(NPAD * KGEMM + 255) / 256, 256, 0, stream>>>(w_pred, Bt);
    k_angle<<<BATCH, 256, 0, stream>>>(points, w_start, b_start, w_id, b_id,
                                       w_pool, b_pool, w_out, rot);
    k_warp_conv<<<BATCH, 256, 0, stream>>>(images, conv_w, conv_b, rot, reps);
    k_gemm<<<16, 32, 0, stream>>>(reps, Bt, b_pred, out);
}
